// MistralAttentionQ_4380866642366
// MI455X (gfx1250) — matrix operand order repaired and checked by reading, not compiled, not run on hardware
//
#include <hip/hip_runtime.h>
#include <hip/hip_bf16.h>

// ---------------- problem constants ----------------
constexpr int Sdim = 2048;
constexpr int Hdim = 4096;
constexpr int NQ   = 32;
constexpr int NKV  = 8;
constexpr int Dh   = 128;
constexpr float SCALING = 0.08838834764831845f;  // 1/sqrt(128)
constexpr float LOG2E   = 1.4426950408889634f;

// stacked weight rows: [wq(4096) ; wk(1024) ; wv(1024) ; wo(4096)] x H
constexpr size_t WQ_OFF = 0;
constexpr size_t WK_OFF = (size_t)4096 * Hdim;
constexpr size_t WV_OFF = (size_t)5120 * Hdim;
constexpr size_t WO_OFF = (size_t)6144 * Hdim;
constexpr size_t W_ELEMS = (size_t)10240 * Hdim;          // 41,943,040
constexpr size_t HS_ELEMS = (size_t)Sdim * Hdim;          //  8,388,608
constexpr size_t Q_ELEMS  = (size_t)NQ * Sdim * Dh;       //  8,388,608
constexpr size_t K_ELEMS  = (size_t)NKV * Sdim * Dh;      //  2,097,152

typedef __attribute__((ext_vector_type(16))) __bf16 v16bf;
typedef __attribute__((ext_vector_type(8)))  __bf16 v8bf;
typedef __attribute__((ext_vector_type(8)))  float  v8f;

// ---------------- WMMA helpers ----------------
__device__ inline v8f wmma_bf16(v16bf a, v16bf b, v8f c) {
  return __builtin_amdgcn_wmma_f32_16x16x32_bf16(
      false, a, false, b, (short)0, c, false, false);
}
// split-precision product: (ah+al)*(bh+bl) ~= ah*bh + ah*bl + al*bh
__device__ inline v8f wmma3(v16bf ah, v16bf al, v16bf bh, v16bf bl, v8f c) {
  c = wmma_bf16(ah, bh, c);
  c = wmma_bf16(ah, bl, c);
  c = wmma_bf16(al, bh, c);
  return c;
}

// A fragment: two aligned runs of 8 bf16 -> v16bf (pure loads, no VALU)
__device__ inline v16bf loadA(const __bf16* p0, const __bf16* p1) {
  v8bf a = *reinterpret_cast<const v8bf*>(p0);
  v8bf b = *reinterpret_cast<const v8bf*>(p1);
  return __builtin_shufflevector(a, b, 0, 1, 2, 3, 4, 5, 6, 7,
                                       8, 9, 10, 11, 12, 13, 14, 15);
}
// B fragment: two aligned runs of 8 bf16 at p and p+16 -> v16bf
__device__ inline v16bf loadB(const __bf16* p) {
  return loadA(p, p + 16);
}

__device__ inline void splitStore(float v, __bf16* hi, __bf16* lo, size_t i) {
  __bf16 h = (__bf16)v;
  hi[i] = h;
  lo[i] = (__bf16)(v - (float)h);
}

// fp32 (LDS staging, stride-33) -> bf16 hi/lo A-fragment (softmax P only)
__device__ inline void split16s(const float* r0, const float* r1,
                                v16bf& hi, v16bf& lo) {
  float v[16];
#pragma unroll
  for (int e = 0; e < 8; ++e) { v[e] = r0[e]; v[8 + e] = r1[e]; }
#pragma unroll
  for (int e = 0; e < 16; ++e) {
    __bf16 h = (__bf16)v[e];
    hi[e] = h;
    lo[e] = (__bf16)(v[e] - (float)h);
  }
}

__device__ inline float redmax16(float v) {
#pragma unroll
  for (int o = 1; o < 16; o <<= 1) v = fmaxf(v, __shfl_xor(v, o, 32));
  return v;
}
__device__ inline float redsum16(float v) {
#pragma unroll
  for (int o = 1; o < 16; o <<= 1) v += __shfl_xor(v, o, 32);
  return v;
}

// ============== pre-pass: split weights into bf16 hi/lo planes ==============
__global__ __launch_bounds__(256)
void cvt_w_k(const float* __restrict__ wq, const float* __restrict__ wk,
             const float* __restrict__ wv, const float* __restrict__ wo,
             __bf16* __restrict__ whi, __bf16* __restrict__ wlo) {
  const size_t i = (size_t)blockIdx.x * 256 + threadIdx.x;  // < W_ELEMS
  float v;
  if (i < WK_OFF)      v = wq[i];
  else if (i < WV_OFF) v = wk[i - WK_OFF];
  else if (i < WO_OFF) v = wv[i - WV_OFF];
  else                 v = wo[i - WO_OFF];
  splitStore(v, whi, wlo, i);
}

// ============== pre-pass: split hidden_states ==============
__global__ __launch_bounds__(256)
void cvt_hs_k(const float* __restrict__ hs,
              __bf16* __restrict__ hhi, __bf16* __restrict__ hlo) {
  const size_t i = (size_t)blockIdx.x * 256 + threadIdx.x;  // < HS_ELEMS
  splitStore(hs[i], hhi, hlo, i);
}

// ============== Kernel: fused QKV projection ==============
// grid (24,128), block 128 (4 waves). Wave: 16 rows x 64 cols.
// cols [0,4096)->qraw fp32; [4096,5120)->kraw fp32; [5120,6144)->V hi/lo
// planes stored transposed [h][d][s].
__global__ __launch_bounds__(128)
void qkv_gemm_k(const __bf16* __restrict__ hhi, const __bf16* __restrict__ hlo,
                const __bf16* __restrict__ whi, const __bf16* __restrict__ wlo,
                float* __restrict__ qraw, float* __restrict__ kraw,
                __bf16* __restrict__ vhi, __bf16* __restrict__ vlo) {
  const int lane = threadIdx.x & 31, wid = threadIdx.x >> 5;
  const int hf = lane >> 4, m = lane & 15;
  const int rowBase = blockIdx.y * 16;
  const int colBase = blockIdx.x * 256 + wid * 64;  // global col / weight row

  const v8f z = {0.f,0.f,0.f,0.f,0.f,0.f,0.f,0.f};
  v8f acc[4];
#pragma unroll
  for (int n = 0; n < 4; ++n) acc[n] = z;

  const size_t aOff = (size_t)(rowBase + m) * Hdim;
  for (int k0 = 0; k0 < Hdim; k0 += 32) {
    const v16bf ahi = loadA(hhi + aOff + k0 + hf * 8,
                            hhi + aOff + k0 + 16 + hf * 8);
    const v16bf alo = loadA(hlo + aOff + k0 + hf * 8,
                            hlo + aOff + k0 + 16 + hf * 8);
#pragma unroll
    for (int n = 0; n < 4; ++n) {
      const size_t bOff = (size_t)(colBase + n * 16 + m) * Hdim + k0 + hf * 8;
      acc[n] = wmma3(ahi, alo, loadB(whi + bOff), loadB(wlo + bOff), acc[n]);
    }
  }

#pragma unroll
  for (int n = 0; n < 4; ++n) {
#pragma unroll
    for (int r = 0; r < 8; ++r) {
      const int row = rowBase + r + 8 * hf;
      const int col = colBase + n * 16 + m;
      const float val = acc[n][r];
      if (col < 4096) {
        qraw[((size_t)(col >> 7) * Sdim + row) * Dh + (col & 127)] = val;
      } else if (col < 5120) {
        const int cc = col - 4096;
        kraw[((size_t)(cc >> 7) * Sdim + row) * Dh + (cc & 127)] = val;
      } else {
        const int cc = col - 5120;
        splitStore(val, vhi, vlo,
                   ((size_t)(cc >> 7) * Dh + (cc & 127)) * Sdim + row);
      }
    }
  }
}

// ============== Kernel: RoPE -> bf16 hi/lo planes ==============
// one thread per (head, s, d<64) pair; race-free (reads its own pair only).
__global__ __launch_bounds__(256)
void rope_k(const float* __restrict__ qraw, const float* __restrict__ kraw,
            const float* __restrict__ cosb, const float* __restrict__ sinb,
            __bf16* __restrict__ qhi, __bf16* __restrict__ qlo,
            __bf16* __restrict__ khi, __bf16* __restrict__ klo) {
  const int idx  = blockIdx.x * 256 + threadIdx.x;  // < 40*2048*64
  const int d    = idx & 63;
  const int s    = (idx >> 6) & (Sdim - 1);
  const int head = idx >> 17;
  const bool isQ = head < NQ;
  const size_t base = isQ ? ((size_t)head * Sdim + s) * Dh
                          : ((size_t)(head - NQ) * Sdim + s) * Dh;
  const float* src = isQ ? qraw : kraw;
  __bf16* hi = isQ ? qhi : khi;
  __bf16* lo = isQ ? qlo : klo;
  const float x1 = src[base + d], x2 = src[base + d + 64];
  const float c1 = cosb[s * Dh + d],      s1 = sinb[s * Dh + d];
  const float c2 = cosb[s * Dh + d + 64], s2 = sinb[s * Dh + d + 64];
  splitStore(x1 * c1 - x2 * s1, hi, lo, base + d);
  splitStore(x2 * c2 + x1 * s2, hi, lo, base + d + 64);
}

// ============== Kernel: causal flash attention ==============
// one wave per (head, 16-query tile). grid 1024, block 128 (4 waves).
__global__ __launch_bounds__(128)
void attn_k(const __bf16* __restrict__ qhi, const __bf16* __restrict__ qlo,
            const __bf16* __restrict__ khi, const __bf16* __restrict__ klo,
            const __bf16* __restrict__ vhi, const __bf16* __restrict__ vlo,
            __bf16* __restrict__ ahi, __bf16* __restrict__ alo) {
  __shared__ float pbuf[4][16 * 33];
  const int lane = threadIdx.x & 31, wid = threadIdx.x >> 5;
  const int gid   = blockIdx.x * 4 + wid;  // 0..4095
  const int h     = gid >> 7;
  const int qBase = (gid & 127) * 16;
  const int hk    = h >> 2;                // GQA 4:1
  const int hf = lane >> 4, m = lane & 15;

  // persistent Q fragments: 4 K-chunks of 32, hi/lo planes
  v16bf fqh[4], fql[4];
  const size_t qOff = ((size_t)h * Sdim + qBase + m) * Dh;
#pragma unroll
  for (int c = 0; c < 4; ++c) {
    fqh[c] = loadA(qhi + qOff + c * 32 + hf * 8,
                   qhi + qOff + c * 32 + 16 + hf * 8);
    fql[c] = loadA(qlo + qOff + c * 32 + hf * 8,
                   qlo + qOff + c * 32 + 16 + hf * 8);
  }

  const v8f z = {0.f,0.f,0.f,0.f,0.f,0.f,0.f,0.f};
  v8f o[8];
#pragma unroll
  for (int dc = 0; dc < 8; ++dc) o[dc] = z;
  float mrow[8], lrow[8];
#pragma unroll
  for (int r = 0; r < 8; ++r) { mrow[r] = -3.0e38f; lrow[r] = 0.f; }

  float* pb = pbuf[wid];

  for (int kb = 0; kb < qBase + 16; kb += 32) {
    // ---- scores: two 16x16 tiles over 32 keys ----
    v8f s0 = z, s1 = z;
#pragma unroll
    for (int c = 0; c < 4; ++c) {
      const size_t k0 = ((size_t)hk * Sdim + kb + m) * Dh + c * 32 + hf * 8;
      s0 = wmma3(fqh[c], fql[c], loadB(khi + k0), loadB(klo + k0), s0);
      const size_t k1 = k0 + (size_t)16 * Dh;
      s1 = wmma3(fqh[c], fql[c], loadB(khi + k1), loadB(klo + k1), s1);
    }

    // ---- online softmax over the 32-key block ----
#pragma unroll
    for (int r = 0; r < 8; ++r) {
      const int row  = r + 8 * hf;
      const int qIdx = qBase + row;
      float a0 = s0[r] * SCALING;
      float a1 = s1[r] * SCALING;
      if (kb + m      > qIdx) a0 = -1.0e30f;  // causal
      if (kb + 16 + m > qIdx) a1 = -1.0e30f;
      const float bm   = redmax16(fmaxf(a0, a1));
      const float newm = fmaxf(mrow[r], bm);
      const float scal = exp2f((mrow[r] - newm) * LOG2E);
      const float p0 = exp2f((a0 - newm) * LOG2E);
      const float p1 = exp2f((a1 - newm) * LOG2E);
      lrow[r] = lrow[r] * scal + redsum16(p0 + p1);
      mrow[r] = newm;
#pragma unroll
      for (int dc = 0; dc < 8; ++dc) o[dc][r] *= scal;
      pb[row * 33 + m]      = p0;   // stage P for A-layout repack
      pb[row * 33 + 16 + m] = p1;
    }
    asm volatile("s_wait_dscnt 0" ::: "memory");  // LDS write->read, in-wave

    v16bf phi, plo;
    const float* pRow = pb + m * 33;
    split16s(pRow + hf * 8, pRow + 16 + hf * 8, phi, plo);

    // ---- O += P(16x32) @ V(32x128) ----
#pragma unroll
    for (int dc = 0; dc < 8; ++dc) {
      const size_t v0 = ((size_t)hk * Dh + dc * 16 + m) * Sdim + kb + hf * 8;
      o[dc] = wmma3(phi, plo, loadB(vhi + v0), loadB(vlo + v0), o[dc]);
    }
    asm volatile("s_wait_dscnt 0" ::: "memory");  // reads done before reuse
  }

  // ---- normalize + store per-head output as hi/lo planes ----
#pragma unroll
  for (int dc = 0; dc < 8; ++dc) {
#pragma unroll
    for (int r = 0; r < 8; ++r) {
      const int row = r + 8 * hf;
      splitStore(o[dc][r] / lrow[r], ahi, alo,
                 ((size_t)h * Sdim + qBase + row) * Dh + dc * 16 + m);
    }
  }
}

// ============== Kernel: O projection ==============
__global__ __launch_bounds__(128)
void oproj_k(const __bf16* __restrict__ ahi, const __bf16* __restrict__ alo,
             const __bf16* __restrict__ whi, const __bf16* __restrict__ wlo,
             float* __restrict__ out) {
  const int lane = threadIdx.x & 31, wid = threadIdx.x >> 5;
  const int hf = lane >> 4, m = lane & 15;
  const int rowBase = blockIdx.y * 16;
  const int colBase = blockIdx.x * 256 + wid * 64;

  const v8f z = {0.f,0.f,0.f,0.f,0.f,0.f,0.f,0.f};
  v8f acc[4];
#pragma unroll
  for (int n = 0; n < 4; ++n) acc[n] = z;

  for (int k0 = 0; k0 < Hdim; k0 += 32) {
    // k index j = head*128 + d; a 32-chunk never crosses a head boundary
    const size_t aOff = ((size_t)(k0 >> 7) * Sdim + rowBase + m) * Dh
                      + (k0 & 127);
    const v16bf fah = loadA(ahi + aOff + hf * 8, ahi + aOff + 16 + hf * 8);
    const v16bf fal = loadA(alo + aOff + hf * 8, alo + aOff + 16 + hf * 8);
#pragma unroll
    for (int n = 0; n < 4; ++n) {
      const size_t bOff = WO_OFF
          + (size_t)(colBase + n * 16 + m) * Hdim + k0 + hf * 8;
      acc[n] = wmma3(fah, fal, loadB(whi + bOff), loadB(wlo + bOff), acc[n]);
    }
  }

#pragma unroll
  for (int n = 0; n < 4; ++n) {
#pragma unroll
    for (int r = 0; r < 8; ++r) {
      const int row = rowBase + r + 8 * hf;
      out[(size_t)row * Hdim + colBase + n * 16 + m] = acc[n][r];
    }
  }
}

// ============== host launcher ==============
extern "C" void kernel_launch(void* const* d_in, const int* in_sizes, int n_in,
                              void* d_out, int out_size, void* d_ws,
                              size_t ws_size, hipStream_t stream) {
  const float* hs   = (const float*)d_in[0];
  const float* cosb = (const float*)d_in[1];
  const float* sinb = (const float*)d_in[2];
  // d_in[3] = attention_mask (causal, applied analytically) — unused
  const float* wq = (const float*)d_in[4];
  const float* wk = (const float*)d_in[5];
  const float* wv = (const float*)d_in[6];
  const float* wo = (const float*)d_in[7];
  float* out = (float*)d_out;

  // ---- workspace layout ----
  char* p = (char*)d_ws;
  float* qraw = (float*)p;            p += Q_ELEMS * sizeof(float);
  float* kraw = (float*)p;            p += K_ELEMS * sizeof(float);
  __bf16* hhi = (__bf16*)p;           p += HS_ELEMS * sizeof(__bf16);
  __bf16* hlo = (__bf16*)p;           p += HS_ELEMS * sizeof(__bf16);
  __bf16* whi = (__bf16*)p;           p += W_ELEMS * sizeof(__bf16);
  __bf16* wlo = (__bf16*)p;           p += W_ELEMS * sizeof(__bf16);
  __bf16* qhi = (__bf16*)p;           p += Q_ELEMS * sizeof(__bf16);
  __bf16* qlo = (__bf16*)p;           p += Q_ELEMS * sizeof(__bf16);
  __bf16* khi = (__bf16*)p;           p += K_ELEMS * sizeof(__bf16);
  __bf16* klo = (__bf16*)p;           p += K_ELEMS * sizeof(__bf16);
  __bf16* vhi = (__bf16*)p;           p += K_ELEMS * sizeof(__bf16);
  __bf16* vlo = (__bf16*)p;           p += K_ELEMS * sizeof(__bf16);
  __bf16* ahi = (__bf16*)p;           p += Q_ELEMS * sizeof(__bf16);
  __bf16* alo = (__bf16*)p;

  // 1) split weights / activations into bf16 hi/lo planes (bandwidth-bound)
  cvt_w_k<<<(unsigned)(W_ELEMS / 256), 256, 0, stream>>>(wq, wk, wv, wo,
                                                         whi, wlo);
  cvt_hs_k<<<(unsigned)(HS_ELEMS / 256), 256, 0, stream>>>(hs, hhi, hlo);
  // 2) fused QKV projection (V written transposed as hi/lo planes)
  qkv_gemm_k<<<dim3(24, 128), 128, 0, stream>>>(hhi, hlo, whi, wlo,
                                                qraw, kraw, vhi, vlo);
  // 3) RoPE -> q/k hi/lo planes: (32+8)*2048*64 threads
  rope_k<<<20480, 256, 0, stream>>>(qraw, kraw, cosb, sinb,
                                    qhi, qlo, khi, klo);
  // 4) causal flash attention: 32 heads * 128 q-tiles = 4096 waves
  attn_k<<<1024, 128, 0, stream>>>(qhi, qlo, khi, klo, vhi, vlo, ahi, alo);
  // 5) output projection
  oproj_k<<<dim3(16, 128), 128, 0, stream>>>(ahi, alo, whi, wlo, out);
}
